// SignedGraphConvolutionalNetwork_46213848105917
// MI455X (gfx1250) — matrix operand order repaired and checked by reading, not compiled, not run on hardware
//
#include <hip/hip_runtime.h>
#include <hip/hip_bf16.h>
#include <math.h>

typedef __attribute__((ext_vector_type(16))) _Float16 v16h;
typedef __attribute__((ext_vector_type(8)))  _Float16 v8h;
typedef __attribute__((ext_vector_type(8)))  float    v8f;
typedef int v4i __attribute__((vector_size(16)));   // b128 payload type for async-LDS builtins

#define NNODES 4096
#define DDIM   256
#define EEDGES 131072

#define AS1 __attribute__((address_space(1)))
#define AS3 __attribute__((address_space(3)))

#if __has_builtin(__builtin_amdgcn_global_load_async_to_lds_b128) && \
    __has_builtin(__builtin_amdgcn_s_wait_asynccnt)
#define USE_ASYNC 1
#else
#define USE_ASYNC 0
#endif

// B tile staged in LDS: 64 rows (cols of C) x 32 f16 (one k-step) = 64 B data
// per row, padded to 80 B to spread LDS banks. Double buffered.
#define LDS_STRIDE 80
#define LDS_BUF    (64 * LDS_STRIDE)   // 5120 B per buffer

// ---------------------------------------------------------------- utilities
__global__ void zero_f32_k(float* p, long n) {
  long i = (long)blockIdx.x * blockDim.x + threadIdx.x;
  if (i < n) p[i] = 0.0f;
}
__global__ void copy_f32_k(const float* __restrict__ a, float* __restrict__ b, long n) {
  long i = (long)blockIdx.x * blockDim.x + threadIdx.x;
  if (i < n) b[i] = a[i];
}
__global__ void f32_to_f16_k(const float* __restrict__ a, _Float16* __restrict__ b, long n) {
  long i = (long)blockIdx.x * blockDim.x + threadIdx.x;
  if (i < n) b[i] = (_Float16)a[i];
}
// in: [K, Nout] row-major f32 -> out: [Nout, K] row-major f16 (B^T for WMMA B-frag loads)
__global__ void transpose_to_f16_k(const float* __restrict__ in, _Float16* __restrict__ out,
                                   int K, int Nout) {
  long t = (long)blockIdx.x * blockDim.x + threadIdx.x;
  if (t >= (long)K * Nout) return;
  int n = (int)(t / K), k = (int)(t - (long)n * K);
  out[t] = (_Float16)in[(size_t)k * Nout + n];
}

// ---------------------------------------------------------------- graph ops
__global__ void deg_k(const int* __restrict__ rows, float* __restrict__ deg) {
  int i = blockIdx.x * blockDim.x + threadIdx.x;
  if (i < EEDGES) atomicAdd(&deg[rows[i]], 1.0f);
}
// agg[rows[e], :] += feat[cols[e], :]   (float4 per thread)
__global__ void scatter_add_k(const float* __restrict__ feat, const int* __restrict__ rows,
                              const int* __restrict__ cols, int F, float* __restrict__ agg) {
  long t = (long)blockIdx.x * blockDim.x + threadIdx.x;
  int per = F >> 2;
  int e = (int)(t / per);
  if (e >= EEDGES) return;
  int j = (int)(t - (long)e * per) << 2;
  int r = rows[e], c = cols[e];
  const float4 v = *(const float4*)(feat + (size_t)c * F + j);
  float* dst = agg + (size_t)r * F + j;
  atomicAdd(dst + 0, v.x); atomicAdd(dst + 1, v.y);
  atomicAdd(dst + 2, v.z); atomicAdd(dst + 3, v.w);
}

// cat = f16([agg/deg , X1])  (base layer, width 2*DDIM)
__global__ void build_cat_base_k(const float* __restrict__ agg, const float* __restrict__ deg,
                                 const float* __restrict__ X1, _Float16* __restrict__ cat) {
  const int W = 2 * DDIM;
  long t = (long)blockIdx.x * blockDim.x + threadIdx.x;
  int i = (int)(t / W);
  if (i >= NNODES) return;
  int j = (int)(t - (long)i * W);
  float v = (j < DDIM) ? agg[(size_t)i * DDIM + j] / fmaxf(deg[i], 1.0f)
                       : X1[(size_t)i * DDIM + (j - DDIM)];
  cat[t] = (_Float16)v;
}
// cat = f16([o1/(d1+1), o2/(d2+1), h])  (deep layers, F=128, width 384)
__global__ void build_cat_deep_k(const float* __restrict__ o1, const float* __restrict__ d1,
                                 const float* __restrict__ o2, const float* __restrict__ d2,
                                 const float* __restrict__ h, _Float16* __restrict__ cat) {
  const int F = 128, W = 384;
  long t = (long)blockIdx.x * blockDim.x + threadIdx.x;
  int i = (int)(t / W);
  if (i >= NNODES) return;
  int j = (int)(t - (long)i * W);
  float v;
  if (j < F)          v = o1[(size_t)i * F + j]           / (d1[i] + 1.0f);
  else if (j < 2 * F) v = o2[(size_t)i * F + (j - F)]     / (d2[i] + 1.0f);
  else                v = h[(size_t)i * F + (j - 2 * F)];
  cat[t] = (_Float16)v;
}

// -------------------------------------------------------- WMMA tile mainloop
// Per-wave 16x64 tile of C. B tile (64 cols x 32 k) staged in LDS once per
// block (8 waves share it), double-buffered with async global->LDS copies.
// blockDim must be 256. All threads participate (barriers inside).
__device__ __forceinline__ void wmma_tile_16x64(
    const _Float16* __restrict__ arow,   // A + (row0 + m) * K
    const _Float16* __restrict__ Bt,     // [Ncols, K] row-major
    int K, int col0, char* lds, v8f acc[4]) {
  const int tid  = threadIdx.x;
  const int lane = tid & 31;
  const int m    = lane & 15;
  const int half = lane >> 4;
  // copy role: 256 threads x 16 B = one 64x32-f16 tile per k-step
  const int crow = tid >> 2;           // 0..63
  const int cseg = tid & 3;            // 0..3 (16-byte segment)
  const _Float16* gsrc0 = Bt + (size_t)(col0 + crow) * K + cseg * 8;
  char* ldst0 = lds + crow * LDS_STRIDE + cseg * 16;
  const int nk = K >> 5;

#if USE_ASYNC
  __builtin_amdgcn_global_load_async_to_lds_b128((AS1 v4i*)gsrc0, (AS3 v4i*)ldst0, 0, 0);
  __builtin_amdgcn_s_wait_asynccnt(0);
#else
  *(float4*)ldst0 = *(const float4*)gsrc0;
#endif
  __syncthreads();

  for (int kk = 0; kk < nk; ++kk) {
    const int cur = kk & 1;
    const int k0  = kk << 5;
#if USE_ASYNC
    if (kk + 1 < nk) {  // prefetch next B tile into the other buffer
      __builtin_amdgcn_global_load_async_to_lds_b128(
          (AS1 v4i*)(gsrc0 + k0 + 32), (AS3 v4i*)(ldst0 + (1 - cur) * LDS_BUF), 0, 0);
    }
#endif
    union { v16h v; v8h h[2]; } af;
    af.h[0] = *(const v8h*)(arow + k0 + half * 8);
    af.h[1] = *(const v8h*)(arow + k0 + 16 + half * 8);
    const char* bbase = lds + cur * LDS_BUF;
#pragma unroll
    for (int t = 0; t < 4; ++t) {
      const char* bp = bbase + (t * 16 + m) * LDS_STRIDE + half * 16;
      union { v16h v; v8h h[2]; } bf;
      bf.h[0] = *(const v8h*)(bp);
      bf.h[1] = *(const v8h*)(bp + 32);
      acc[t] = __builtin_amdgcn_wmma_f32_16x16x32_f16(false, af.v, false, bf.v,
                                                      (short)0, acc[t], false, false);
    }
    if (kk + 1 < nk) {
#if USE_ASYNC
      __builtin_amdgcn_s_wait_asynccnt(0);   // my prefetch landed
      __syncthreads();                       // everyone's landed; reads of cur done
#else
      __syncthreads();                       // reads of other buffer done everywhere
      *(float4*)(ldst0 + (1 - cur) * LDS_BUF) = *(const float4*)(gsrc0 + k0 + 32);
      __syncthreads();
#endif
    }
  }
}

// ---------------------------------------------------------------- WMMA GEMM
// out[M,Nout] = A[M,K](f16) @ Bt[Nout,K](f16)^T + bias, optional relu.
__global__ void gemm_wmma_k(const _Float16* __restrict__ A, const _Float16* __restrict__ Bt,
                            const float* __restrict__ bias, float* __restrict__ out,
                            int M, int K, int Nout, int relu) {
  __shared__ alignas(16) char ldsb[2 * LDS_BUF];
  const int lane = threadIdx.x & 31;
  const int wave = threadIdx.x >> 5;
  const int row0 = (blockIdx.y * 8 + wave) * 16;
  const int col0 = blockIdx.x * 64;
  const int m = lane & 15, half = lane >> 4;

  v8f acc[4] = {v8f{}, v8f{}, v8f{}, v8f{}};
  wmma_tile_16x64(A + (size_t)(row0 + m) * K, Bt, K, col0, ldsb, acc);

#pragma unroll
  for (int t = 0; t < 4; ++t) {
    const int col = col0 + t * 16 + m;
    const float b = bias[col];
#pragma unroll
    for (int v = 0; v < 8; ++v) {
      const int row = row0 + v + half * 8;
      float val = acc[t][v] + b;
      if (relu) val = fmaxf(val, 0.0f);
      out[(size_t)row * Nout + col] = val;
    }
  }
}

// ------------------------------------------------- row l2-normalize (+tanh)
__global__ void rownorm_k(const float* __restrict__ in, float* __restrict__ out,
                          _Float16* __restrict__ outh, int F, int do_tanh) {
  __shared__ float red[128];
  const int i = blockIdx.x, j = threadIdx.x;
  float v = (j < F) ? in[(size_t)i * F + j] : 0.0f;
  red[j] = v * v;
  __syncthreads();
  for (int s = 64; s > 0; s >>= 1) {
    if (j < s) red[j] += red[j + s];
    __syncthreads();
  }
  const float inv = 1.0f / fmaxf(sqrtf(red[0]), 1e-12f);
  if (j < F) {
    float y = v * inv;
    if (do_tanh) y = tanhf(y);
    out[(size_t)i * F + j] = y;
    if (outh) outh[(size_t)i * F + j] = (_Float16)y;
  }
}

__global__ void concat2_k(const float* __restrict__ a, const float* __restrict__ b,
                          float* __restrict__ z, int F) {
  long t = (long)blockIdx.x * blockDim.x + threadIdx.x;
  int i = (int)(t / (2 * F));
  if (i >= NNODES) return;
  int j = (int)(t - (long)i * 2 * F);
  z[t] = (j < F) ? a[(size_t)i * F + j] : b[(size_t)i * F + (j - F)];
}

// ---------------------- fused X_mol @ X_mol^T * mask, pred store, MSE reduce
__global__ void simgemm_k(const _Float16* __restrict__ Xh, const float* __restrict__ mask,
                          const float* __restrict__ labels, float* __restrict__ pred,
                          float* __restrict__ lossAcc) {
  __shared__ alignas(16) char ldsb[2 * LDS_BUF];
  const int K = 128;
  const int lane = threadIdx.x & 31;
  const int wave = threadIdx.x >> 5;
  const int row0 = (blockIdx.y * 8 + wave) * 16;
  const int col0 = blockIdx.x * 64;
  const int m = lane & 15, half = lane >> 4;

  v8f acc[4] = {v8f{}, v8f{}, v8f{}, v8f{}};
  // column n of B == row n of X_mol (both row-major f16)
  wmma_tile_16x64(Xh + (size_t)(row0 + m) * K, Xh, K, col0, ldsb, acc);

  float lsum = 0.0f;
#pragma unroll
  for (int t = 0; t < 4; ++t) {
    const int col = col0 + t * 16 + m;
#pragma unroll
    for (int v = 0; v < 8; ++v) {
      const int row = row0 + v + half * 8;
      const size_t idx = (size_t)row * NNODES + col;
      const float p = acc[t][v] * mask[idx];
      pred[idx] = p;
      const float d = p - labels[idx];
      lsum += d * d;
    }
  }
  __shared__ float red[256];
  red[threadIdx.x] = lsum;
  __syncthreads();
  for (int s = 128; s > 0; s >>= 1) {
    if ((int)threadIdx.x < s) red[threadIdx.x] += red[threadIdx.x + s];
    __syncthreads();
  }
  if (threadIdx.x == 0) atomicAdd(lossAcc, red[0]);
}

__global__ void finalize_k(const float* __restrict__ acc, float* __restrict__ out) {
  out[0] = acc[0] * (1.0f / ((float)NNODES * (float)NNODES));
}

// ---------------------------------------------------------------- launcher
extern "C" void kernel_launch(void* const* d_in, const int* in_sizes, int n_in,
                              void* d_out, int out_size, void* d_ws, size_t ws_size,
                              hipStream_t stream) {
  (void)in_sizes; (void)n_in; (void)out_size; (void)ws_size;

  const float* X      = (const float*)d_in[0];
  const int*   pe     = (const int*)d_in[1];   // [2,E]: rows=pe, cols=pe+E
  const int*   ne     = (const int*)d_in[2];
  const float* labels = (const float*)d_in[3];
  const float* mask   = (const float*)d_in[4];
  const float* lin_w  = (const float*)d_in[5];
  const float* lin_b  = (const float*)d_in[6];
  const float* pb_w   = (const float*)d_in[7];
  const float* pb_b   = (const float*)d_in[8];
  const float* nb_w   = (const float*)d_in[9];
  const float* nb_b   = (const float*)d_in[10];
  const float* pd_w1  = (const float*)d_in[11];
  const float* pd_b1  = (const float*)d_in[12];
  const float* nd_w1  = (const float*)d_in[13];
  const float* nd_b1  = (const float*)d_in[14];
  const float* pd_w2  = (const float*)d_in[15];
  const float* pd_b2  = (const float*)d_in[16];
  const float* nd_w2  = (const float*)d_in[17];
  const float* nd_b2  = (const float*)d_in[18];

  float* out_loss = (float*)d_out;
  float* out_xmol = out_loss + 1;
  float* out_pred = out_xmol + (size_t)NNODES * 128;

  char* base = (char*)d_ws;
  size_t off = 0;
  auto alloc = [&](size_t bytes) -> void* {
    void* p = base + off;
    off = (off + bytes + 255) & ~(size_t)255;
    return p;
  };
  _Float16* Xh    = (_Float16*)alloc((size_t)NNODES * DDIM * 2);
  _Float16* WtLin = (_Float16*)alloc((size_t)DDIM * DDIM * 2);
  _Float16* WtPb  = (_Float16*)alloc((size_t)128 * 512 * 2);
  _Float16* WtNb  = (_Float16*)alloc((size_t)128 * 512 * 2);
  _Float16* WtPd1 = (_Float16*)alloc((size_t)128 * 384 * 2);
  _Float16* WtNd1 = (_Float16*)alloc((size_t)128 * 384 * 2);
  _Float16* WtPd2 = (_Float16*)alloc((size_t)64 * 384 * 2);
  _Float16* WtNd2 = (_Float16*)alloc((size_t)64 * 384 * 2);
  float* X1   = (float*)alloc((size_t)NNODES * DDIM * 4);
  float* degP = (float*)alloc((size_t)NNODES * 4);
  float* degN = (float*)alloc((size_t)NNODES * 4);
  float* agg  = (float*)alloc((size_t)NNODES * DDIM * 4);
  _Float16* cat = (_Float16*)alloc((size_t)NNODES * 512 * 2);
  float* rawP = (float*)alloc((size_t)NNODES * 128 * 4);
  float* rawN = (float*)alloc((size_t)NNODES * 128 * 4);
  float* hP   = (float*)alloc((size_t)NNODES * 128 * 4);
  float* hN   = (float*)alloc((size_t)NNODES * 128 * 4);
  float* hP2  = (float*)alloc((size_t)NNODES * 128 * 4);
  float* hN2  = (float*)alloc((size_t)NNODES * 128 * 4);
  float* oA   = (float*)alloc((size_t)NNODES * 128 * 4);
  float* oB   = (float*)alloc((size_t)NNODES * 128 * 4);
  float* oC   = (float*)alloc((size_t)NNODES * 128 * 4);
  float* oD   = (float*)alloc((size_t)NNODES * 128 * 4);
  float* z    = (float*)alloc((size_t)NNODES * 128 * 4);
  _Float16* Xmolh = (_Float16*)alloc((size_t)NNODES * 128 * 2);
  float* lossAcc  = (float*)alloc(4);

  const int T = 256;
  auto cdiv = [](long a, long b) { return (int)((a + b - 1) / b); };
  auto gemm = [&](const _Float16* A, const _Float16* Bt, const float* bias, float* out,
                  int M, int K, int Nout, int relu) {
    dim3 g(Nout / 64, M / 128);
    gemm_wmma_k<<<g, 256, 0, stream>>>(A, Bt, bias, out, M, K, Nout, relu);
  };

  // init / conversions
  zero_f32_k<<<1, 32, 0, stream>>>(lossAcc, 1);
  zero_f32_k<<<cdiv(NNODES, T), T, 0, stream>>>(degP, NNODES);
  zero_f32_k<<<cdiv(NNODES, T), T, 0, stream>>>(degN, NNODES);
  f32_to_f16_k<<<cdiv((long)NNODES * DDIM, T), T, 0, stream>>>(X, Xh, (long)NNODES * DDIM);
  transpose_to_f16_k<<<cdiv((long)DDIM * DDIM, T), T, 0, stream>>>(lin_w, WtLin, DDIM, DDIM);
  transpose_to_f16_k<<<cdiv(512L * 128, T), T, 0, stream>>>(pb_w, WtPb, 512, 128);
  transpose_to_f16_k<<<cdiv(512L * 128, T), T, 0, stream>>>(nb_w, WtNb, 512, 128);
  transpose_to_f16_k<<<cdiv(384L * 128, T), T, 0, stream>>>(pd_w1, WtPd1, 384, 128);
  transpose_to_f16_k<<<cdiv(384L * 128, T), T, 0, stream>>>(nd_w1, WtNd1, 384, 128);
  transpose_to_f16_k<<<cdiv(384L * 64, T), T, 0, stream>>>(pd_w2, WtPd2, 384, 64);
  transpose_to_f16_k<<<cdiv(384L * 64, T), T, 0, stream>>>(nd_w2, WtNd2, 384, 64);
  deg_k<<<cdiv(EEDGES, T), T, 0, stream>>>(pe, degP);
  deg_k<<<cdiv(EEDGES, T), T, 0, stream>>>(ne, degN);

  // X1 = relu(X @ lin_w + lin_b)
  gemm(Xh, WtLin, lin_b, X1, NNODES, DDIM, DDIM, 1);

  // base layer, positive sign
  zero_f32_k<<<cdiv((long)NNODES * DDIM, T), T, 0, stream>>>(agg, (long)NNODES * DDIM);
  scatter_add_k<<<cdiv((long)EEDGES * (DDIM / 4), T), T, 0, stream>>>(X1, pe, pe + EEDGES, DDIM, agg);
  build_cat_base_k<<<cdiv((long)NNODES * 2 * DDIM, T), T, 0, stream>>>(agg, degP, X1, cat);
  gemm(cat, WtPb, pb_b, rawP, NNODES, 512, 128, 0);
  rownorm_k<<<NNODES, 128, 0, stream>>>(rawP, hP, (_Float16*)nullptr, 128, 1);

  // base layer, negative sign
  zero_f32_k<<<cdiv((long)NNODES * DDIM, T), T, 0, stream>>>(agg, (long)NNODES * DDIM);
  scatter_add_k<<<cdiv((long)EEDGES * (DDIM / 4), T), T, 0, stream>>>(X1, ne, ne + EEDGES, DDIM, agg);
  build_cat_base_k<<<cdiv((long)NNODES * 2 * DDIM, T), T, 0, stream>>>(agg, degN, X1, cat);
  gemm(cat, WtNb, nb_b, rawN, NNODES, 512, 128, 0);
  rownorm_k<<<NNODES, 128, 0, stream>>>(rawN, hN, (_Float16*)nullptr, 128, 1);

  // deep layers (self-loops: copy-init sums, divide by deg+1)
  auto deep = [&](const float* xP, const float* xN,
                  const _Float16* WtP, const float* bP,
                  const _Float16* WtN, const float* bN,
                  int Fout, float* outP, float* outN) {
    const int F = 128;
    const long nf = (long)NNODES * F;
    copy_f32_k<<<cdiv(nf, T), T, 0, stream>>>(xP, oA, nf);
    copy_f32_k<<<cdiv(nf, T), T, 0, stream>>>(xN, oB, nf);
    copy_f32_k<<<cdiv(nf, T), T, 0, stream>>>(xN, oC, nf);
    copy_f32_k<<<cdiv(nf, T), T, 0, stream>>>(xP, oD, nf);
    const int sg = cdiv((long)EEDGES * (F / 4), T);
    scatter_add_k<<<sg, T, 0, stream>>>(xP, pe, pe + EEDGES, F, oA);
    scatter_add_k<<<sg, T, 0, stream>>>(xN, ne, ne + EEDGES, F, oB);
    scatter_add_k<<<sg, T, 0, stream>>>(xN, pe, pe + EEDGES, F, oC);
    scatter_add_k<<<sg, T, 0, stream>>>(xP, ne, ne + EEDGES, F, oD);
    build_cat_deep_k<<<cdiv((long)NNODES * 384, T), T, 0, stream>>>(oA, degP, oB, degN, xP, cat);
    gemm(cat, WtP, bP, rawP, NNODES, 384, Fout, 0);
    build_cat_deep_k<<<cdiv((long)NNODES * 384, T), T, 0, stream>>>(oC, degP, oD, degN, xN, cat);
    gemm(cat, WtN, bN, rawN, NNODES, 384, Fout, 0);
    rownorm_k<<<NNODES, 128, 0, stream>>>(rawP, outP, (_Float16*)nullptr, Fout, 1);
    rownorm_k<<<NNODES, 128, 0, stream>>>(rawN, outN, (_Float16*)nullptr, Fout, 1);
  };
  deep(hP, hN, WtPd1, pd_b1, WtNd1, nd_b1, 128, hP2, hN2);
  deep(hP2, hN2, WtPd2, pd_b2, WtNd2, nd_b2, 64, hP, hN);   // hP/hN now [N,64]

  // X_mol = l2norm([hP, hN])
  concat2_k<<<cdiv((long)NNODES * 128, T), T, 0, stream>>>(hP, hN, z, 64);
  rownorm_k<<<NNODES, 128, 0, stream>>>(z, out_xmol, Xmolh, 128, 0);

  // pred = (X_mol @ X_mol^T) * mask ; loss = mean((pred - labels)^2)
  dim3 gs(NNODES / 64, NNODES / 128);
  simgemm_k<<<gs, 256, 0, stream>>>(Xmolh, mask, labels, out_pred, lossAcc);
  finalize_k<<<1, 1, 0, stream>>>(lossAcc, out_loss);
}
